// Seq2SeqLSTM_80041010528335
// MI455X (gfx1250) — matrix operand order repaired and checked by reading, not compiled, not run on hardware
//
#include <hip/hip_runtime.h>
#include <hip/hip_bf16.h>

// ---------------------------------------------------------------------------
// Seq2Seq LSTM for MI455X (gfx1250, wave32, WMMA), fused step kernel.
//
// Per (timestep, layer): one kernel computes
//   z = A1 @ W1^T + A2 @ W2^T          (bf16 WMMA, f32 accum)
//   i,f,g,o gates + c/h update         (fused epilogue, z stays in VGPRs)
//
// Partition: grid.x = H/16 = 64 blocks; block = 256 thr = 8 waves.
//   wave w  -> M rows 16w..16w+15 (batch)
//   block j0 = 16*blockIdx.x -> columns {g*H + j0 + 0..15} for g = i,f,g,o
// Each wave holds 4 accumulators (one per gate). One A-tile load feeds 4
// v_wmma_f32_16x16x32_bf16 ops. After K-loop, each lane owns all four gate
// pre-activations for its 8 (row,col) elements -> pure-VALU LSTM epilogue.
//
// h is double-buffered (kernel reads h_prev, writes h_next); c is in-place
// (each element owned by exactly one lane). Weights (~51MB bf16) stay
// L2-resident (192MB L2).
// ---------------------------------------------------------------------------

#define B_  128
#define T_  512
#define IN_ 64
#define H_  1024
#define S_  64
#define N4H 4096   // 4*H

typedef __attribute__((ext_vector_type(16))) __bf16 v16bf;
typedef __attribute__((ext_vector_type(8)))  float  v8f;

union V16B { v16bf v; uint4 q[2]; };

__device__ __forceinline__ float sigmoidf_(float x)
{
    return 1.f / (1.f + __expf(-x));
}

// VGPR layouts (bf16 16x16x32); A/B K order is this code's contiguous choice, C/D per CDNA5 ISA 7.12.2:
//  A (16x32): lane<16 -> M=lane, elems[0..15]=K0..15;
//             lane>=16 -> elems[0..15]=K16..31
//  B (32x16): lane<16 -> N=lane, elems[0..15]=K0..15; lane>=16 -> K16..31
//  C/D f32:   vgpr r: lane<16 -> (M=r,N=lane); lane>=16 -> (M=r+8,N=lane-16)
__global__ __launch_bounds__(256) void lstm_step_fused(
    const __bf16* __restrict__ A1, int lda1, int K1, const __bf16* __restrict__ W1,
    const __bf16* __restrict__ A2, int lda2, int K2, const __bf16* __restrict__ W2,
    const float* __restrict__ bcomb,                 // 4H combined bias
    const float* __restrict__ xin,                   // B, optional rank-1 input
    const float* __restrict__ wcol,                  // 4H f32, optional
    float* __restrict__ c,                           // B*H, in-place
    __bf16* __restrict__ hout)                       // B*H, h_next buffer
{
    const int lane = threadIdx.x & 31;
    const int wave = threadIdx.x >> 5;    // 0..7 -> M tile
    const int j0   = blockIdx.x * 16;     // column slice within H
    const int half = lane >> 4;           // 0 or 1
    const int l15  = lane & 15;

    v8f acc[4] = {};                      // i, f, g, o gate tiles

    if (K1 > 0) {
        const __bf16* __restrict__ arow = A1 + (size_t)(wave * 16 + l15) * (size_t)lda1;
        const __bf16* __restrict__ w0 = W1 + (size_t)(0 * H_ + j0 + l15) * (size_t)K1;
        const __bf16* __restrict__ w1 = W1 + (size_t)(1 * H_ + j0 + l15) * (size_t)K1;
        const __bf16* __restrict__ w2 = W1 + (size_t)(2 * H_ + j0 + l15) * (size_t)K1;
        const __bf16* __restrict__ w3 = W1 + (size_t)(3 * H_ + j0 + l15) * (size_t)K1;
        for (int k0 = 0; k0 < K1; k0 += 32) {
            V16B a, b0, b1, b2, b3;
            const __bf16* ap = arow + k0 + 16 * half;
            a.q[0] = *(const uint4*)(ap);
            a.q[1] = *(const uint4*)(ap + 8);
            const int bo = k0 + 16 * half;
            b0.q[0] = *(const uint4*)(w0 + bo); b0.q[1] = *(const uint4*)(w0 + bo + 8);
            b1.q[0] = *(const uint4*)(w1 + bo); b1.q[1] = *(const uint4*)(w1 + bo + 8);
            b2.q[0] = *(const uint4*)(w2 + bo); b2.q[1] = *(const uint4*)(w2 + bo + 8);
            b3.q[0] = *(const uint4*)(w3 + bo); b3.q[1] = *(const uint4*)(w3 + bo + 8);
            acc[0] = __builtin_amdgcn_wmma_f32_16x16x32_bf16(false, a.v, false, b0.v, (short)0, acc[0], false, false);
            acc[1] = __builtin_amdgcn_wmma_f32_16x16x32_bf16(false, a.v, false, b1.v, (short)0, acc[1], false, false);
            acc[2] = __builtin_amdgcn_wmma_f32_16x16x32_bf16(false, a.v, false, b2.v, (short)0, acc[2], false, false);
            acc[3] = __builtin_amdgcn_wmma_f32_16x16x32_bf16(false, a.v, false, b3.v, (short)0, acc[3], false, false);
        }
    }
    if (K2 > 0) {
        const __bf16* __restrict__ arow = A2 + (size_t)(wave * 16 + l15) * (size_t)lda2;
        const __bf16* __restrict__ w0 = W2 + (size_t)(0 * H_ + j0 + l15) * (size_t)K2;
        const __bf16* __restrict__ w1 = W2 + (size_t)(1 * H_ + j0 + l15) * (size_t)K2;
        const __bf16* __restrict__ w2 = W2 + (size_t)(2 * H_ + j0 + l15) * (size_t)K2;
        const __bf16* __restrict__ w3 = W2 + (size_t)(3 * H_ + j0 + l15) * (size_t)K2;
        for (int k0 = 0; k0 < K2; k0 += 32) {
            V16B a, b0, b1, b2, b3;
            const __bf16* ap = arow + k0 + 16 * half;
            a.q[0] = *(const uint4*)(ap);
            a.q[1] = *(const uint4*)(ap + 8);
            const int bo = k0 + 16 * half;
            b0.q[0] = *(const uint4*)(w0 + bo); b0.q[1] = *(const uint4*)(w0 + bo + 8);
            b1.q[0] = *(const uint4*)(w1 + bo); b1.q[1] = *(const uint4*)(w1 + bo + 8);
            b2.q[0] = *(const uint4*)(w2 + bo); b2.q[1] = *(const uint4*)(w2 + bo + 8);
            b3.q[0] = *(const uint4*)(w3 + bo); b3.q[1] = *(const uint4*)(w3 + bo + 8);
            acc[0] = __builtin_amdgcn_wmma_f32_16x16x32_bf16(false, a.v, false, b0.v, (short)0, acc[0], false, false);
            acc[1] = __builtin_amdgcn_wmma_f32_16x16x32_bf16(false, a.v, false, b1.v, (short)0, acc[1], false, false);
            acc[2] = __builtin_amdgcn_wmma_f32_16x16x32_bf16(false, a.v, false, b2.v, (short)0, acc[2], false, false);
            acc[3] = __builtin_amdgcn_wmma_f32_16x16x32_bf16(false, a.v, false, b3.v, (short)0, acc[3], false, false);
        }
    }

    // ---- fused LSTM epilogue (PyTorch gate order i,f,g,o) ----
    const int col     = j0 + l15;
    const int rowbase = wave * 16 + 8 * half;

    const float bi = bcomb[col];
    const float bf = bcomb[H_ + col];
    const float bg = bcomb[2 * H_ + col];
    const float bo = bcomb[3 * H_ + col];

    const bool hasx = (xin != nullptr);
    float wi = 0.f, wf = 0.f, wg = 0.f, wo = 0.f;
    if (hasx) {
        wi = wcol[col];
        wf = wcol[H_ + col];
        wg = wcol[2 * H_ + col];
        wo = wcol[3 * H_ + col];
    }

    #pragma unroll
    for (int r = 0; r < 8; ++r) {
        const int row = rowbase + r;
        float zi = acc[0][r] + bi;
        float zf = acc[1][r] + bf;
        float zg = acc[2][r] + bg;
        float zo = acc[3][r] + bo;
        if (hasx) {
            const float xv = xin[row];
            zi += xv * wi; zf += xv * wf; zg += xv * wg; zo += xv * wo;
        }
        const float ig = sigmoidf_(zi);
        const float fg = sigmoidf_(zf);
        const float gv = tanhf(zg);
        const float og = sigmoidf_(zo);
        const size_t idx = (size_t)row * H_ + col;
        const float cn = fg * c[idx] + ig * gv;
        c[idx]    = cn;
        hout[idx] = (__bf16)(og * tanhf(cn));
    }
}

// out[b, s] = dot(h1[b,:], fcW) + fcb ; also feeds decoder input buffer.
__global__ __launch_bounds__(256) void fc_out(
    const __bf16* __restrict__ h1, const float* __restrict__ fcW,
    const float* __restrict__ fcb, float* __restrict__ outg,
    float* __restrict__ inp, int s)
{
    const int b = blockIdx.x;
    float sum = 0.f;
    for (int j = threadIdx.x; j < H_; j += 256)
        sum += (float)h1[(size_t)b * H_ + j] * fcW[j];
    __shared__ float red[256];
    red[threadIdx.x] = sum;
    __syncthreads();
    for (int o = 128; o > 0; o >>= 1) {
        if (threadIdx.x < o) red[threadIdx.x] += red[threadIdx.x + o];
        __syncthreads();
    }
    if (threadIdx.x == 0) {
        const float v = red[0] + fcb[0];
        outg[(size_t)b * S_ + s] = v;
        inp[b] = v;
    }
}

// ------------------------------ setup helpers ------------------------------
__global__ void f32_to_bf16(const float* __restrict__ s, __bf16* __restrict__ d, int n)
{
    int i = blockIdx.x * blockDim.x + threadIdx.x;
    if (i < n) d[i] = (__bf16)s[i];
}
__global__ void bias_combine(const float* __restrict__ a, const float* __restrict__ b,
                             float* __restrict__ o, int n)
{
    int i = blockIdx.x * blockDim.x + threadIdx.x;
    if (i < n) o[i] = a[i] + b[i];
}
__global__ void fill_zero_f32(float* p, int n)
{
    int i = blockIdx.x * blockDim.x + threadIdx.x;
    if (i < n) p[i] = 0.f;
}
__global__ void fill_zero_bf16(__bf16* p, int n)
{
    int i = blockIdx.x * blockDim.x + threadIdx.x;
    if (i < n) p[i] = (__bf16)0.f;
}

// ---------------------------------------------------------------------------
extern "C" void kernel_launch(void* const* d_in, const int* in_sizes, int n_in,
                              void* d_out, int out_size, void* d_ws, size_t ws_size,
                              hipStream_t stream)
{
    (void)in_sizes; (void)n_in; (void)out_size; (void)ws_size;

    const float* x        = (const float*)d_in[0];
    const float* eWih0f   = (const float*)d_in[1];
    const float* eWhh0f   = (const float*)d_in[2];
    const float* ebih0    = (const float*)d_in[3];
    const float* ebhh0    = (const float*)d_in[4];
    const float* eWih1f   = (const float*)d_in[5];
    const float* eWhh1f   = (const float*)d_in[6];
    const float* ebih1    = (const float*)d_in[7];
    const float* ebhh1    = (const float*)d_in[8];
    const float* dWih0f   = (const float*)d_in[9];   // (4H,1): rank-1 term in epilogue
    const float* dWhh0f   = (const float*)d_in[10];
    const float* dbih0    = (const float*)d_in[11];
    const float* dbhh0    = (const float*)d_in[12];
    const float* dWih1f   = (const float*)d_in[13];
    const float* dWhh1f   = (const float*)d_in[14];
    const float* dbih1    = (const float*)d_in[15];
    const float* dbhh1    = (const float*)d_in[16];
    const float* fcW      = (const float*)d_in[17];
    const float* fcb      = (const float*)d_in[18];
    float*       outg     = (float*)d_out;

    // ----- workspace layout (256B aligned chunks) -----
    char* ws = (char*)d_ws;
    size_t off = 0;
    auto take = [&](size_t bytes) -> char* {
        char* p = ws + off;
        off = (off + bytes + 255) & ~(size_t)255;
        return p;
    };
    __bf16* xbf   = (__bf16*)take((size_t)B_ * T_ * IN_ * 2);
    __bf16* eWih0 = (__bf16*)take((size_t)N4H * IN_ * 2);
    __bf16* eWhh0 = (__bf16*)take((size_t)N4H * H_ * 2);
    __bf16* eWih1 = (__bf16*)take((size_t)N4H * H_ * 2);
    __bf16* eWhh1 = (__bf16*)take((size_t)N4H * H_ * 2);
    __bf16* dWhh0 = (__bf16*)take((size_t)N4H * H_ * 2);
    __bf16* dWih1 = (__bf16*)take((size_t)N4H * H_ * 2);
    __bf16* dWhh1 = (__bf16*)take((size_t)N4H * H_ * 2);
    float*  bc_e0 = (float*)take(N4H * 4);
    float*  bc_e1 = (float*)take(N4H * 4);
    float*  bc_d0 = (float*)take(N4H * 4);
    float*  bc_d1 = (float*)take(N4H * 4);
    __bf16* h0a   = (__bf16*)take((size_t)B_ * H_ * 2);   // h double buffers
    __bf16* h0b   = (__bf16*)take((size_t)B_ * H_ * 2);
    __bf16* h1a   = (__bf16*)take((size_t)B_ * H_ * 2);
    __bf16* h1b   = (__bf16*)take((size_t)B_ * H_ * 2);
    float*  c0    = (float*)take((size_t)B_ * H_ * 4);
    float*  c1    = (float*)take((size_t)B_ * H_ * 4);
    float*  inp   = (float*)take(B_ * 4);

    const int BH = B_ * H_;            // 131072
    auto nb = [](int n) { return (n + 255) / 256; };

    // ----- conversions / init (re-run each call: deterministic) -----
    f32_to_bf16<<<nb(B_ * T_ * IN_), 256, 0, stream>>>(x, xbf, B_ * T_ * IN_);
    f32_to_bf16<<<nb(N4H * IN_), 256, 0, stream>>>(eWih0f, eWih0, N4H * IN_);
    f32_to_bf16<<<nb(N4H * H_), 256, 0, stream>>>(eWhh0f, eWhh0, N4H * H_);
    f32_to_bf16<<<nb(N4H * H_), 256, 0, stream>>>(eWih1f, eWih1, N4H * H_);
    f32_to_bf16<<<nb(N4H * H_), 256, 0, stream>>>(eWhh1f, eWhh1, N4H * H_);
    f32_to_bf16<<<nb(N4H * H_), 256, 0, stream>>>(dWhh0f, dWhh0, N4H * H_);
    f32_to_bf16<<<nb(N4H * H_), 256, 0, stream>>>(dWih1f, dWih1, N4H * H_);
    f32_to_bf16<<<nb(N4H * H_), 256, 0, stream>>>(dWhh1f, dWhh1, N4H * H_);
    bias_combine<<<nb(N4H), 256, 0, stream>>>(ebih0, ebhh0, bc_e0, N4H);
    bias_combine<<<nb(N4H), 256, 0, stream>>>(ebih1, ebhh1, bc_e1, N4H);
    bias_combine<<<nb(N4H), 256, 0, stream>>>(dbih0, dbhh0, bc_d0, N4H);
    bias_combine<<<nb(N4H), 256, 0, stream>>>(dbih1, dbhh1, bc_d1, N4H);
    fill_zero_bf16<<<nb(BH), 256, 0, stream>>>(h0a, BH);
    fill_zero_bf16<<<nb(BH), 256, 0, stream>>>(h1a, BH);
    fill_zero_f32<<<nb(BH), 256, 0, stream>>>(c0, BH);
    fill_zero_f32<<<nb(BH), 256, 0, stream>>>(c1, BH);

    __bf16 *h0c = h0a, *h0n = h0b;     // current / next
    __bf16 *h1c = h1a, *h1n = h1b;

    const int GRID = H_ / 16;          // 64 blocks

    // ----- encoder: interleave layer0/layer1 per timestep -----
    for (int t = 0; t < T_; ++t) {
        lstm_step_fused<<<GRID, 256, 0, stream>>>(
            xbf + (size_t)t * IN_, T_ * IN_, IN_, eWih0,
            h0c, H_, H_, eWhh0,
            bc_e0, nullptr, nullptr, c0, h0n);
        lstm_step_fused<<<GRID, 256, 0, stream>>>(
            h0n, H_, H_, eWih1,
            h1c, H_, H_, eWhh1,
            bc_e1, nullptr, nullptr, c1, h1n);
        { __bf16* tmp = h0c; h0c = h0n; h0n = tmp; }
        { __bf16* tmp = h1c; h1c = h1n; h1n = tmp; }
    }

    // ----- autoregressive decoder (continues from encoder finals) -----
    for (int s = 0; s < S_; ++s) {
        lstm_step_fused<<<GRID, 256, 0, stream>>>(
            nullptr, 0, 0, nullptr,
            h0c, H_, H_, dWhh0,
            bc_d0,
            (s == 0) ? nullptr : inp,          // inp0 == zeros -> skip term
            (s == 0) ? nullptr : dWih0f,
            c0, h0n);
        lstm_step_fused<<<GRID, 256, 0, stream>>>(
            h0n, H_, H_, dWih1,
            h1c, H_, H_, dWhh1,
            bc_d1, nullptr, nullptr, c1, h1n);
        fc_out<<<B_, 256, 0, stream>>>(h1n, fcW, fcb, outg, inp, s);
        { __bf16* tmp = h0c; h0c = h0n; h0n = tmp; }
        { __bf16* tmp = h1c; h1c = h1n; h1n = tmp; }
    }
}
